// ImprovedTaskAllocator_35983236006576
// MI455X (gfx1250) — matrix operand order repaired and checked by reading, not compiled, not run on hardware
//
#include <hip/hip_runtime.h>

typedef _Float16 h16;
typedef __attribute__((ext_vector_type(16))) _Float16 v16h;
typedef __attribute__((ext_vector_type(8)))  float    v8f;

#define HD 128
#define NHEAD 4
#define NLAYER 2
#define FFD 256
#define BB 512
#define RR 50
#define TT 20

// ---------------------------------------------------------------- weight prep
__global__ void cvt_f16_kernel(const float* __restrict__ src, h16* __restrict__ dst, int n) {
  int i = blockIdx.x * 256 + threadIdx.x;
  if (i < n) dst[i] = (h16)src[i];
}

// alloc_w1 is [128, 256]; split into task half [128,128] and robot half [128,128]
__global__ void split_w1_kernel(const float* __restrict__ w1,
                                h16* __restrict__ w1a, h16* __restrict__ w1b) {
  int i = blockIdx.x * 256 + threadIdx.x;
  if (i < 128 * 128) {
    int r = i >> 7, k = i & 127;
    w1a[i] = (h16)w1[r * 256 + k];
    w1b[i] = (h16)w1[r * 256 + 128 + k];
  }
}

// ---------------------------------------------------------------- input projection (K = 7 or 6)
__global__ void proj_kernel(const float* __restrict__ st, const float* __restrict__ W,
                            const float* __restrict__ bias, float* __restrict__ X32,
                            h16* __restrict__ X16, int M, int KD) {
  int i = blockIdx.x * 256 + threadIdx.x;
  if (i >= M * HD) return;
  int m = i >> 7, n = i & 127;
  float acc = bias[n];
  const float* s = st + (size_t)m * KD;
  const float* w = W + (size_t)n * KD;
  for (int k = 0; k < KD; ++k) acc += s[k] * w[k];
  X32[i] = acc;
  X16[i] = (h16)acc;
}

// ---------------------------------------------------------------- WMMA fragment helpers
// 16-bit A 16x32 wave32 layout: lanes 0-15 = rows, v0..3 -> K 0..7 (+8 for hi lanes),
// v4..7 -> K 16..23 (+8 for hi lanes)
__device__ __forceinline__ v16h load_a_frag(const h16* __restrict__ row, int k0, bool hi) {
  v16h a;
#pragma unroll
  for (int v = 0; v < 8; ++v) {
    int ka = k0 + ((v < 4) ? 0 : 16) + (hi ? 8 : 0) + ((v & 3) << 1);
    a[2 * v]     = row[ka];
    a[2 * v + 1] = row[ka + 1];
  }
  return a;
}
// B 32x16 wave32 layout: lanes 0-15 = cols, v0..3 -> K 0..7 (+8 for hi lanes), v4..7 -> K 16..23 (+8 for hi lanes)
__device__ __forceinline__ v16h load_b_frag(const h16* __restrict__ row, int k0, bool hi) {
  v16h b;
#pragma unroll
  for (int v = 0; v < 8; ++v) {
    int kb = k0 + ((v < 4) ? 0 : 16) + (hi ? 8 : 0) + ((v & 3) << 1);
    b[2 * v]     = row[kb];
    b[2 * v + 1] = row[kb + 1];
  }
  return b;
}
// C/D 32-bit 16x16 layout: lanes 0-15 -> N, VGPR r -> M=r (+8 for hi lanes)
__device__ __forceinline__ void store_frag(v8f c, int row0, int col, int N, bool hi,
                                           float bv, int flags,
                                           float* __restrict__ out32, h16* __restrict__ out16) {
#pragma unroll
  for (int r = 0; r < 8; ++r) {
    int row = row0 + r + (hi ? 8 : 0);
    float val = c[r] + bv;
    if (flags & 2) val = fmaxf(val, 0.f);
    size_t idx = (size_t)row * N + col;
    if (flags & 4) out32[idx] = val;
    if (flags & 8) out16[idx] = (h16)val;
  }
}

// ---------------------------------------------------------------- WMMA GEMM (32x32 tile per wave)
// Y[M,N] = A[M,K](f16, row-major) x W[N,K](f16, row-major, used transposed) (+bias)(+relu)
// 2x2 register blocking: 4 WMMAs per 8 x b128 fragment loads per K-step of 32.
// flags: 1 = add bias, 2 = relu, 4 = store fp32, 8 = store fp16
__global__ void gemm_wmma_kernel(const h16* __restrict__ A, const h16* __restrict__ W,
                                 const float* __restrict__ bias,
                                 float* __restrict__ out32, h16* __restrict__ out16,
                                 int M, int N, int K, int flags) {
  int wid  = blockIdx.x * 8 + (threadIdx.x >> 5);
  int lane = threadIdx.x & 31;
  int tilesM = M >> 5;
  int tiles  = tilesM * (N >> 5);
  if (wid >= tiles) return;                 // wave-uniform: EXEC stays all-1s for WMMA
  int tm = wid % tilesM, tn = wid / tilesM;
  bool hi = lane >= 16;
  int  l15 = lane & 15;

  const h16* arow0 = A + (size_t)(tm * 32 + l15) * K;
  const h16* arow1 = arow0 + (size_t)16 * K;
  const h16* wrow0 = W + (size_t)(tn * 32 + l15) * K;
  const h16* wrow1 = wrow0 + (size_t)16 * K;

  v8f c00 = {}, c01 = {}, c10 = {}, c11 = {};
  for (int k0 = 0; k0 < K; k0 += 32) {
    v16h a0 = load_a_frag(arow0, k0, hi);
    v16h a1 = load_a_frag(arow1, k0, hi);
    v16h b0 = load_b_frag(wrow0, k0, hi);
    v16h b1 = load_b_frag(wrow1, k0, hi);
    c00 = __builtin_amdgcn_wmma_f32_16x16x32_f16(false, a0, false, b0, (short)0, c00, false, false);
    c01 = __builtin_amdgcn_wmma_f32_16x16x32_f16(false, a0, false, b1, (short)0, c01, false, false);
    c10 = __builtin_amdgcn_wmma_f32_16x16x32_f16(false, a1, false, b0, (short)0, c10, false, false);
    c11 = __builtin_amdgcn_wmma_f32_16x16x32_f16(false, a1, false, b1, (short)0, c11, false, false);
  }

  int col0 = tn * 32 + l15;
  int col1 = col0 + 16;
  float bv0 = (flags & 1) ? bias[col0] : 0.f;
  float bv1 = (flags & 1) ? bias[col1] : 0.f;
  int row0 = tm * 32;
  store_frag(c00, row0,      col0, N, hi, bv0, flags, out32, out16);
  store_frag(c01, row0,      col1, N, hi, bv1, flags, out32, out16);
  store_frag(c10, row0 + 16, col0, N, hi, bv0, flags, out32, out16);
  store_frag(c11, row0 + 16, col1, N, hi, bv1, flags, out32, out16);
}

// ---------------------------------------------------------------- attention (per batch, per head)
template <int S>
__global__ void attn_kernel(const float* __restrict__ qkv, h16* __restrict__ out16) {
  int b  = blockIdx.x >> 2;   // / NHEAD
  int hh = blockIdx.x & 3;
  int tid = threadIdx.x;      // 0..63
  const float* base = qkv + (size_t)b * S * (3 * HD);

  __shared__ float Ks[S][32];
  __shared__ float Vs[S][32];
  __shared__ float Ps[64][(S | 1)];   // odd stride -> bank-conflict-free rows

  for (int i = tid; i < S * 32; i += 64) {
    int s = i >> 5, d = i & 31;
    Ks[s][d] = base[(size_t)s * (3 * HD) + HD + hh * 32 + d];
    Vs[s][d] = base[(size_t)s * (3 * HD) + 2 * HD + hh * 32 + d];
  }
  __syncthreads();

  if (tid < S) {
    float q[32];
#pragma unroll
    for (int d = 0; d < 32; ++d) q[d] = base[(size_t)tid * (3 * HD) + hh * 32 + d];
    const float scale = 0.17677669529663687f;   // 1/sqrt(32)
    float mx = -1e30f;
    for (int j = 0; j < S; ++j) {
      float acc = 0.f;
#pragma unroll
      for (int d = 0; d < 32; ++d) acc += q[d] * Ks[j][d];
      acc *= scale;
      Ps[tid][j] = acc;
      mx = fmaxf(mx, acc);
    }
    float sum = 0.f;
    for (int j = 0; j < S; ++j) { float e = __expf(Ps[tid][j] - mx); Ps[tid][j] = e; sum += e; }
    float inv = 1.f / sum;
    float o[32];
#pragma unroll
    for (int d = 0; d < 32; ++d) o[d] = 0.f;
    for (int j = 0; j < S; ++j) {
      float p = Ps[tid][j];
#pragma unroll
      for (int d = 0; d < 32; ++d) o[d] += p * Vs[j][d];
    }
    h16* op = out16 + (size_t)(b * S + tid) * HD + hh * 32;
#pragma unroll
    for (int d = 0; d < 32; ++d) op[d] = (h16)(o[d] * inv);
  }
}

// ---------------------------------------------------------------- residual + layernorm (wave32 per row)
__global__ void add_ln_kernel(float* __restrict__ X, const float* __restrict__ Y,
                              const float* __restrict__ g, const float* __restrict__ bb,
                              h16* __restrict__ X16, int M) {
  int row  = blockIdx.x * 8 + (threadIdx.x >> 5);
  int lane = threadIdx.x & 31;
  if (row >= M) return;
  size_t rb = (size_t)row * HD;
  float v[4];
  float s = 0.f;
#pragma unroll
  for (int i = 0; i < 4; ++i) {
    int cc = lane * 4 + i;
    v[i] = X[rb + cc] + Y[rb + cc];
    s += v[i];
  }
#pragma unroll
  for (int off = 16; off >= 1; off >>= 1) s += __shfl_xor(s, off, 32);
  float mean = s * (1.f / 128.f);
  float var = 0.f;
#pragma unroll
  for (int i = 0; i < 4; ++i) { float d = v[i] - mean; var += d * d; }
#pragma unroll
  for (int off = 16; off >= 1; off >>= 1) var += __shfl_xor(var, off, 32);
  float inv = rsqrtf(var * (1.f / 128.f) + 1e-5f);
#pragma unroll
  for (int i = 0; i < 4; ++i) {
    int cc = lane * 4 + i;
    float o = (v[i] - mean) * inv * g[cc] + bb[cc];
    X[rb + cc]  = o;
    X16[rb + cc] = (h16)o;
  }
}

// ---------------------------------------------------------------- pairwise relu-dot + softmax over R
__global__ void scores_kernel(const float* __restrict__ Ab, const float* __restrict__ Bb,
                              const float* __restrict__ w2, const float* __restrict__ b2,
                              float* __restrict__ out) {
  int bt  = blockIdx.x;          // = b*TT + t
  int b   = bt / TT;
  int tid = threadIdx.x;         // 64 threads
  __shared__ float arow[HD], w2s[HD], sc[64];
  for (int i = tid; i < HD; i += 64) { arow[i] = Ab[(size_t)bt * HD + i]; w2s[i] = w2[i]; }
  __syncthreads();
  float s = -1e30f;
  if (tid < RR) {
    const float* brow = Bb + (size_t)(b * RR + tid) * HD;
    float acc = b2[0];
    for (int k = 0; k < HD; ++k) {
      float h = arow[k] + brow[k];      // a already carries alloc_b1
      h = fmaxf(h, 0.f);
      acc += h * w2s[k];
    }
    s = acc;
  }
  sc[tid] = s;
  __syncthreads();
  float mx = -1e30f;
  for (int j = 0; j < RR; ++j) mx = fmaxf(mx, sc[j]);
  float sum = 0.f;
  for (int j = 0; j < RR; ++j) sum += __expf(sc[j] - mx);
  if (tid < RR) out[(size_t)bt * RR + tid] = __expf(sc[tid] - mx) / sum;
}

// ---------------------------------------------------------------- host
extern "C" void kernel_launch(void* const* d_in, const int* in_sizes, int n_in,
                              void* d_out, int out_size, void* d_ws, size_t ws_size,
                              hipStream_t stream) {
  const float* robot_states = (const float*)d_in[0];
  const float* task_states  = (const float*)d_in[1];
  const float* robot_proj_w = (const float*)d_in[2];
  const float* robot_proj_b = (const float*)d_in[3];
  const float* task_proj_w  = (const float*)d_in[4];
  const float* task_proj_b  = (const float*)d_in[5];
  const float* qkv_w = (const float*)d_in[6];
  const float* qkv_b = (const float*)d_in[7];
  const float* out_w = (const float*)d_in[8];
  const float* out_b = (const float*)d_in[9];
  const float* ff1_w = (const float*)d_in[10];
  const float* ff1_b = (const float*)d_in[11];
  const float* ff2_w = (const float*)d_in[12];
  const float* ff2_b = (const float*)d_in[13];
  const float* ln1_g = (const float*)d_in[14];
  const float* ln1_b = (const float*)d_in[15];
  const float* ln2_g = (const float*)d_in[16];
  const float* ln2_b = (const float*)d_in[17];
  const float* alloc_w1 = (const float*)d_in[18];
  const float* alloc_b1 = (const float*)d_in[19];
  const float* alloc_w2 = (const float*)d_in[20];
  const float* alloc_b2 = (const float*)d_in[21];
  float* out = (float*)d_out;
  (void)in_sizes; (void)n_in; (void)out_size; (void)ws_size;

  char* ws = (char*)d_ws;
  size_t off = 0;
  auto carve = [&](size_t bytes) -> char* {
    char* p = ws + off;
    off += (bytes + 255) & ~(size_t)255;
    return p;
  };
  const int Mr = BB * RR, Mt = BB * TT;
  h16*   qkv16  = (h16*)carve((size_t)4 * 384 * 128 * 2);
  h16*   out16w = (h16*)carve((size_t)4 * 128 * 128 * 2);
  h16*   ff116  = (h16*)carve((size_t)4 * 256 * 128 * 2);
  h16*   ff216  = (h16*)carve((size_t)4 * 128 * 256 * 2);
  h16*   w1a16  = (h16*)carve((size_t)128 * 128 * 2);
  h16*   w1b16  = (h16*)carve((size_t)128 * 128 * 2);
  float* Xr32   = (float*)carve((size_t)Mr * HD * 4);
  float* Xt32   = (float*)carve((size_t)Mt * HD * 4);
  h16*   Xr16   = (h16*)carve((size_t)Mr * HD * 2);
  h16*   Xt16   = (h16*)carve((size_t)Mt * HD * 2);
  float* bufA   = (float*)carve((size_t)Mr * 384 * 4);   // QKV / GEMM fp32 outputs
  h16*   bufB   = (h16*)carve((size_t)Mr * 256 * 2);     // fp16 GEMM inputs (attn out / FF hidden)

  auto cvt = [&](const float* s, h16* d, int n) {
    cvt_f16_kernel<<<(n + 255) / 256, 256, 0, stream>>>(s, d, n);
  };
  cvt(qkv_w, qkv16, 4 * 384 * 128);
  cvt(out_w, out16w, 4 * 128 * 128);
  cvt(ff1_w, ff116, 4 * 256 * 128);
  cvt(ff2_w, ff216, 4 * 128 * 256);
  split_w1_kernel<<<(128 * 128 + 255) / 256, 256, 0, stream>>>(alloc_w1, w1a16, w1b16);

  auto gemm = [&](const h16* A, const h16* W, const float* bias,
                  float* o32, h16* o16, int M, int N, int K, int flags) {
    int tiles = (M / 32) * (N / 32);
    gemm_wmma_kernel<<<(tiles + 7) / 8, 256, 0, stream>>>(A, W, bias, o32, o16, M, N, K, flags);
  };

  for (int e = 0; e < 2; ++e) {
    int    M   = (e == 0) ? Mr : Mt;
    float* X32 = (e == 0) ? Xr32 : Xt32;
    h16*   X16 = (e == 0) ? Xr16 : Xt16;
    if (e == 0)
      proj_kernel<<<(M * HD + 255) / 256, 256, 0, stream>>>(robot_states, robot_proj_w, robot_proj_b, X32, X16, M, 7);
    else
      proj_kernel<<<(M * HD + 255) / 256, 256, 0, stream>>>(task_states, task_proj_w, task_proj_b, X32, X16, M, 6);

    for (int l = 0; l < NLAYER; ++l) {
      int el = e * NLAYER + l;
      // QKV: [M,128] x [384,128]^T -> fp32 [M,384]
      gemm(X16, qkv16 + (size_t)el * 384 * 128, qkv_b + el * 384, bufA, nullptr, M, 384, 128, 1 | 4);
      if (e == 0) attn_kernel<RR><<<BB * NHEAD, 64, 0, stream>>>(bufA, bufB);
      else        attn_kernel<TT><<<BB * NHEAD, 64, 0, stream>>>(bufA, bufB);
      // out proj: fp16 attn [M,128] -> fp32 [M,128]
      gemm(bufB, out16w + (size_t)el * 128 * 128, out_b + el * 128, bufA, nullptr, M, 128, 128, 1 | 4);
      add_ln_kernel<<<M / 8, 256, 0, stream>>>(X32, bufA, ln1_g + el * 128, ln1_b + el * 128, X16, M);
      // FF1 with fused bias+relu, fp16 hidden [M,256]
      gemm(X16, ff116 + (size_t)el * 256 * 128, ff1_b + el * 256, nullptr, bufB, M, 256, 128, 1 | 2 | 8);
      // FF2 -> fp32 [M,128]
      gemm(bufB, ff216 + (size_t)el * 128 * 256, ff2_b + el * 128, bufA, nullptr, M, 128, 256, 1 | 4);
      add_ln_kernel<<<M / 8, 256, 0, stream>>>(X32, bufA, ln2_g + el * 128, ln2_b + el * 128, X16, M);
    }
  }

  // final pairwise scoring: a = tf@W1a^T + b1 ; b = rf@W1b^T ; softmax_r(relu(a+b)@w2 + b2)
  float* Abuf = bufA;
  float* Bbuf = bufA + (size_t)Mt * HD;
  gemm(Xt16, w1a16, alloc_b1, Abuf, nullptr, Mt, 128, 128, 1 | 4);
  gemm(Xr16, w1b16, nullptr,  Bbuf, nullptr, Mr, 128, 128, 4);
  scores_kernel<<<BB * TT, 64, 0, stream>>>(Abuf, Bbuf, alloc_w2, alloc_b2, out);
}
